// MambaV2Block_72361609003495
// MI455X (gfx1250) — matrix operand order repaired and checked by reading, not compiled, not run on hardware
//
#include <hip/hip_runtime.h>

// ---------------- problem constants ----------------
constexpr int cB      = 2;
constexpr int cL      = 2048;
constexpr int cDM     = 1024;               // D_MODEL
constexpr int cDI     = 2048;               // D_INNER
constexpr int cNH     = 32;                 // NHEADS
constexpr int cHD     = 64;                 // HEADDIM
constexpr int cDS     = 128;                // D_STATE
constexpr int cDCONV  = 4;
constexpr int cCONVD  = cDI + 2 * cDS;      // 2304
constexpr int cDPROJ  = 2 * cDI + 2 * cDS + cNH; // 4384
constexpr int cM      = cB * cL;            // 4096 rows
constexpr float cEPS  = 1e-5f;

typedef __attribute__((ext_vector_type(16))) __bf16 v16bf;
typedef __attribute__((ext_vector_type(8)))  __bf16 v8bf;
typedef __attribute__((ext_vector_type(8)))  float  v8f;
typedef unsigned int u32x4 __attribute__((ext_vector_type(4)));
typedef int          i32x4 __attribute__((ext_vector_type(4)));
typedef int          i32x8 __attribute__((ext_vector_type(8)));

#define GLOBAL_AS __attribute__((address_space(1)))
#define LDS_AS    __attribute__((address_space(3)))

#if __has_builtin(__builtin_amdgcn_tensor_load_to_lds)
#define HAVE_TDM 1
#else
#define HAVE_TDM 0
#endif
#if __has_builtin(__builtin_amdgcn_global_load_async_to_lds_b128)
#define HAVE_ASYNC 1
#else
#define HAVE_ASYNC 0
#endif

__device__ __forceinline__ void wait_async0() {
#if __has_builtin(__builtin_amdgcn_s_wait_asynccnt)
    __builtin_amdgcn_s_wait_asynccnt(0);
#else
    asm volatile("s_wait_asynccnt 0x0" ::: "memory");
#endif
}

#if HAVE_ASYNC
__device__ __forceinline__ void async_b128(const __bf16* gsrc, __bf16* ldst) {
    __builtin_amdgcn_global_load_async_to_lds_b128(
        (GLOBAL_AS i32x4*)gsrc, (LDS_AS i32x4*)ldst, 0, 0);
}
#endif

// ---------------- fp32 -> bf16 convert ----------------
__global__ void cvt_f32_bf16_kernel(const float* __restrict__ src,
                                    __bf16* __restrict__ dst, int n) {
    int i = blockIdx.x * 256 + threadIdx.x;
    if (i < n) dst[i] = (__bf16)src[i];
}

// ---------------- LayerNorm -> bf16 ----------------
__global__ __launch_bounds__(256)
void ln_bf16_kernel(const float* __restrict__ x, const float* __restrict__ w,
                    const float* __restrict__ b, __bf16* __restrict__ u) {
    int row = blockIdx.x;
    const float* xr = x + (size_t)row * cDM;
    float lsum = 0.f, lsq = 0.f;
    for (int c = threadIdx.x; c < cDM; c += 256) {
        float v = xr[c];
        lsum += v; lsq += v * v;
    }
    __shared__ float s1[8], s2[8];
    int lane = threadIdx.x & 31, wave = threadIdx.x >> 5;
    for (int off = 16; off > 0; off >>= 1) {
        lsum += __shfl_down(lsum, off, 32);
        lsq  += __shfl_down(lsq,  off, 32);
    }
    if (lane == 0) { s1[wave] = lsum; s2[wave] = lsq; }
    __syncthreads();
    if (threadIdx.x == 0) {
        float a = 0.f, c2 = 0.f;
        for (int i = 0; i < 8; ++i) { a += s1[i]; c2 += s2[i]; }
        s1[0] = a / cDM; s2[0] = c2 / cDM;
    }
    __syncthreads();
    float mu  = s1[0];
    float var = s2[0] - mu * mu;
    float inv = rsqrtf(var + cEPS);
    for (int c = threadIdx.x; c < cDM; c += 256) {
        float v = (xr[c] - mu) * inv * w[c] + b[c];
        u[(size_t)row * cDM + c] = (__bf16)v;
    }
}

// ---------------- WMMA bf16 GEMM:  C[M,N] = A[M,K] * W[N,K]^T (+Res) ----------------
// block tile: 128(M) x 128(N), K step 32; 8 waves (4Mx2N), each wave 2x4 16x16 tiles.
__global__ __launch_bounds__(256)
void gemm_bf16_wmma_kernel(const __bf16* __restrict__ A, const __bf16* __restrict__ W,
                           const float* __restrict__ Res, float* __restrict__ C,
                           int Ndim, int Kdim) {
    __shared__ __bf16 As[128][40];   // padded stride (16 DW data + 4 DW pad)
    __shared__ __bf16 Bs[128][40];

    int t    = threadIdx.x;
    int lane = t & 31, wave = t >> 5;
    int wm = wave & 3, wn = wave >> 2;      // 4 x 2 wave grid
    int mBlock = blockIdx.y * 128;
    int nBlock = blockIdx.x * 128;
    int lm = lane & 15, g = lane >> 4;

    v8f acc[2][4];
#pragma unroll
    for (int i = 0; i < 2; ++i)
#pragma unroll
        for (int j = 0; j < 4; ++j)
#pragma unroll
            for (int r = 0; r < 8; ++r) acc[i][j][r] = 0.f;

    // cooperative staging: each thread moves 16 bf16 of A and 16 bf16 of B
    int sr = t >> 1, sc = (t & 1) * 16;     // 128 rows x 32 cols
    const __bf16* aPtr = A + (size_t)(mBlock + sr) * Kdim + sc;
    int nGlob = nBlock + sr;
    if (nGlob > Ndim - 1) nGlob = Ndim - 1;  // clamp: OOB cols never stored
    const __bf16* wPtr = W + (size_t)nGlob * Kdim + sc;

#if HAVE_TDM
    unsigned ldsA = (unsigned)(size_t)(LDS_AS __bf16*)&As[0][0];
    unsigned long long aBase =
        (unsigned long long)(size_t)A + (unsigned long long)(size_t)mBlock * (unsigned)Kdim * 2ull;
#endif

    for (int k0 = 0; k0 < Kdim; k0 += 32) {
        __syncthreads();
#if HAVE_TDM
        // ---- A tile via Tensor Data Mover (wave 0 issues one 32x128 2D descriptor) ----
        if (wave == 0) {
            unsigned long long ga = aBase + (unsigned)k0 * 2u;
            u32x4 g0; i32x8 g1;
            i32x4 gz = {0, 0, 0, 0};
            g0[0] = 1u;                                   // count=1, user mode
            g0[1] = ldsA;                                 // lds_addr
            g0[2] = (unsigned)(ga & 0xffffffffu);         // global_addr[31:0]
            g0[3] = (unsigned)((ga >> 32) & 0x01ffffffu) | (2u << 30); // addr[56:32], type=2
            // data_size=2B, pad_enable, pad_interval=16DW, pad_amount=4DW
            g1[0] = (1 << 16) | (1 << 20) | (3 << 22) | (3 << 25);
            g1[1] = (int)(((unsigned)Kdim & 0xffffu) << 16);          // tensor_dim0 lo16
            g1[2] = (int)(((unsigned)Kdim >> 16) | ((unsigned)(cM & 0xffff) << 16));
            g1[3] = (int)(((unsigned)cM >> 16) | (32u << 16));        // tile_dim0=32
            g1[4] = 128;                                              // tile_dim1=128
            g1[5] = Kdim;                                             // dim0 stride lo32
            g1[6] = 0;
            g1[7] = 0;
#if defined(__clang_major__) && (__clang_major__ >= 23)
            i32x8 g4 = {0, 0, 0, 0, 0, 0, 0, 0};
            __builtin_amdgcn_tensor_load_to_lds(g0, g1, gz, gz, g4, 0);
#else
            __builtin_amdgcn_tensor_load_to_lds(g0, g1, gz, gz, 0);
#endif
        }
#elif HAVE_ASYNC
        async_b128(aPtr + k0,     &As[sr][sc]);
        async_b128(aPtr + k0 + 8, &As[sr][sc + 8]);
#else
        {
            v8bf a0 = *(const v8bf*)(aPtr + k0);
            v8bf a1 = *(const v8bf*)(aPtr + k0 + 8);
            *(v8bf*)&As[sr][sc]     = a0;
            *(v8bf*)&As[sr][sc + 8] = a1;
        }
#endif
        // ---- B tile ----
#if HAVE_ASYNC
        async_b128(wPtr + k0,     &Bs[sr][sc]);
        async_b128(wPtr + k0 + 8, &Bs[sr][sc + 8]);
#else
        {
            v8bf b0 = *(const v8bf*)(wPtr + k0);
            v8bf b1 = *(const v8bf*)(wPtr + k0 + 8);
            *(v8bf*)&Bs[sr][sc]     = b0;
            *(v8bf*)&Bs[sr][sc + 8] = b1;
        }
#endif
        if (k0 + 32 < Kdim) {       // next-slab prefetch -> global_prefetch_b8
            __builtin_prefetch(aPtr + k0 + 32, 0, 1);
            __builtin_prefetch(wPtr + k0 + 32, 0, 1);
        }
#if HAVE_ASYNC
        wait_async0();
#endif
#if HAVE_TDM
        if (wave == 0) __builtin_amdgcn_s_wait_tensorcnt(0);
#endif
        __syncthreads();

        // A fragment: lane holds row M=lm; lane-half g selects K interleave
        v16bf av[2], bv[4];
#pragma unroll
        for (int i = 0; i < 2; ++i) {
            int m = wm * 32 + i * 16 + lm;
            v8bf lo = *(const v8bf*)&As[m][g * 8];
            v8bf hi = *(const v8bf*)&As[m][16 + g * 8];
            av[i] = __builtin_shufflevector(lo, hi, 0,1,2,3,4,5,6,7,8,9,10,11,12,13,14,15);
        }
        // B fragment: lane holds column N=lm; lane-half g selects K interleave
#pragma unroll
        for (int j = 0; j < 4; ++j) {
            int n = wn * 64 + j * 16 + lm;
            v8bf lo = *(const v8bf*)&Bs[n][g * 8];
            v8bf hi = *(const v8bf*)&Bs[n][16 + g * 8];
            bv[j] = __builtin_shufflevector(lo, hi, 0,1,2,3,4,5,6,7,8,9,10,11,12,13,14,15);
        }
#pragma unroll
        for (int i = 0; i < 2; ++i)
#pragma unroll
            for (int j = 0; j < 4; ++j)
                acc[i][j] = __builtin_amdgcn_wmma_f32_16x16x32_bf16(
                    false, av[i], false, bv[j], (short)0, acc[i][j], false, false);
    }

    // epilogue: D layout -> lane n = lm, rows g*8 + r
#pragma unroll
    for (int i = 0; i < 2; ++i) {
        int rowBase = mBlock + wm * 32 + i * 16 + g * 8;
#pragma unroll
        for (int j = 0; j < 4; ++j) {
            int col = nBlock + wn * 64 + j * 16 + lm;
            if (col < Ndim) {
#pragma unroll
                for (int r = 0; r < 8; ++r) {
                    size_t idx = (size_t)(rowBase + r) * Ndim + col;
                    float v = acc[i][j][r];
                    if (Res) v += Res[idx];
                    C[idx] = v;
                }
            }
        }
    }
}

// ---------------- causal depthwise conv (D_CONV=4) + SiLU ----------------
__global__ void conv_silu_kernel(const float* __restrict__ zx, const float* __restrict__ cw,
                                 const float* __restrict__ cb, float* __restrict__ xconv) {
    long long idx = (long long)blockIdx.x * 256 + threadIdx.x;
    long long total = (long long)cM * cCONVD;
    if (idx >= total) return;
    int c = (int)(idx % cCONVD);
    long long row = idx / cCONVD;
    int l = (int)(row % cL);
    const float* wr = cw + c * cDCONV;
    float acc = cb[c];
#pragma unroll
    for (int j = 0; j < cDCONV; ++j) {
        int ll = l - (cDCONV - 1) + j;
        if (ll >= 0)
            acc += wr[j] * zx[(row + (ll - l)) * cDPROJ + cDI + c];
    }
    float sig = 1.f / (1.f + __expf(-acc));
    xconv[idx] = acc * sig;
}

// ---------------- softplus(dt + dt_bias) ----------------
__global__ void dt_softplus_kernel(const float* __restrict__ zx, const float* __restrict__ dt_bias,
                                   float* __restrict__ dtb) {
    int idx = blockIdx.x * 256 + threadIdx.x;
    if (idx >= cM * cNH) return;
    int hh = idx % cNH;
    int row = idx / cNH;
    float v = zx[(size_t)row * cDPROJ + (2 * cDI + 2 * cDS) + hh] + dt_bias[hh];
    dtb[idx] = (v > 20.f) ? v : log1pf(__expf(v));
}

// ---------------- sequential SSM scan: one block per (batch, head) ----------------
__global__ __launch_bounds__(256)
void ssm_scan_kernel(const float* __restrict__ xconv, const float* __restrict__ dtb,
                     const float* __restrict__ A_log, const float* __restrict__ Dv,
                     float* __restrict__ ybuf) {
    int bh = blockIdx.x;
    int b = bh >> 5, h = bh & 31;
    int t  = threadIdx.x;
    int p  = t >> 2;                 // head-dim index 0..63
    int n0 = (t & 3) * 32;           // this lane's state-dim chunk
    float A  = -__expf(A_log[h]);
    float Dh = Dv[h];
    float hst[32];
#pragma unroll
    for (int i = 0; i < 32; ++i) hst[i] = 0.f;
    __shared__ float bc[256];        // [0..127]=B, [128..255]=C

    for (int l = 0; l < cL; ++l) {
        size_t row = (size_t)b * cL + l;
        const float* xr = xconv + row * cCONVD;
        float dtt = dtb[row * cNH + h];
        float dA  = __expf(dtt * A);
        float xv  = xr[h * cHD + p];
        __syncthreads();
        bc[t] = xr[cDI + t];
        __syncthreads();
        float s  = dtt * xv;
        float yp = 0.f;
#pragma unroll
        for (int i = 0; i < 32; ++i) {
            float Bn = bc[n0 + i];
            float Cn = bc[128 + n0 + i];
            float hv = hst[i] * dA + s * Bn;
            hst[i] = hv;
            yp += hv * Cn;
        }
        yp += __shfl_xor(yp, 1, 32);
        yp += __shfl_xor(yp, 2, 32);
        if ((t & 3) == 0)
            ybuf[row * cDI + h * cHD + p] = yp + Dh * xv;
    }
}

// ---------------- gated RMSNorm -> bf16 ----------------
__global__ __launch_bounds__(256)
void gate_rms_kernel(const float* __restrict__ ybuf, const float* __restrict__ zx,
                     const float* __restrict__ nw, __bf16* __restrict__ gout) {
    int row = blockIdx.x;
    const float* yr = ybuf + (size_t)row * cDI;
    const float* zr = zx + (size_t)row * cDPROJ;   // z = first D_INNER cols
    float gl[8];
    float sq = 0.f;
#pragma unroll
    for (int i = 0; i < 8; ++i) {
        int c = threadIdx.x + i * 256;
        float z = zr[c];
        float sil = z / (1.f + __expf(-z));
        float gv = yr[c] * sil;
        gl[i] = gv;
        sq += gv * gv;
    }
    __shared__ float sred[8];
    int lane = threadIdx.x & 31, wave = threadIdx.x >> 5;
    for (int off = 16; off > 0; off >>= 1) sq += __shfl_down(sq, off, 32);
    if (lane == 0) sred[wave] = sq;
    __syncthreads();
    if (threadIdx.x == 0) {
        float tot = 0.f;
        for (int i = 0; i < 8; ++i) tot += sred[i];
        sred[0] = tot;
    }
    __syncthreads();
    float scale = rsqrtf(sred[0] / cDI + cEPS);
#pragma unroll
    for (int i = 0; i < 8; ++i) {
        int c = threadIdx.x + i * 256;
        gout[(size_t)row * cDI + c] = (__bf16)(gl[i] * scale * nw[c]);
    }
}

// ---------------- launch ----------------
extern "C" void kernel_launch(void* const* d_in, const int* in_sizes, int n_in,
                              void* d_out, int out_size, void* d_ws, size_t ws_size,
                              hipStream_t stream) {
    const float* x        = (const float*)d_in[0];
    const float* ln_w     = (const float*)d_in[1];
    const float* ln_b     = (const float*)d_in[2];
    const float* in_w     = (const float*)d_in[3];
    const float* conv_w   = (const float*)d_in[4];
    const float* conv_b   = (const float*)d_in[5];
    const float* dt_bias  = (const float*)d_in[6];
    const float* A_log    = (const float*)d_in[7];
    const float* Dvec     = (const float*)d_in[8];
    const float* norm_w   = (const float*)d_in[9];
    const float* out_w    = (const float*)d_in[10];
    float* out = (float*)d_out;

    char* ws = (char*)d_ws;
    size_t off = 0;
    auto alloc = [&](size_t bytes) -> void* {
        void* p = ws + off;
        off = (off + bytes + 255) & ~(size_t)255;
        return p;
    };
    __bf16* u_bf   = (__bf16*)alloc((size_t)cM * cDM * 2);
    __bf16* wIn_bf = (__bf16*)alloc((size_t)cDPROJ * cDM * 2);
    __bf16* wOut_bf= (__bf16*)alloc((size_t)cDM * cDI * 2);
    float*  zx     = (float*)alloc((size_t)cM * cDPROJ * 4);
    float*  xconv  = (float*)alloc((size_t)cM * cCONVD * 4);
    float*  dtb    = (float*)alloc((size_t)cM * cNH * 4);
    float*  ybuf   = (float*)alloc((size_t)cM * cDI * 4);
    __bf16* g_bf   = (__bf16*)alloc((size_t)cM * cDI * 2);
    (void)ws_size;

    // weight conversions
    {
        int n1 = cDPROJ * cDM;
        cvt_f32_bf16_kernel<<<(n1 + 255) / 256, 256, 0, stream>>>(in_w, wIn_bf, n1);
        int n2 = cDM * cDI;
        cvt_f32_bf16_kernel<<<(n2 + 255) / 256, 256, 0, stream>>>(out_w, wOut_bf, n2);
    }
    // layernorm
    ln_bf16_kernel<<<cM, 256, 0, stream>>>(x, ln_w, ln_b, u_bf);
    // in_proj GEMM: zx[M, 4384] = u * in_w^T
    {
        dim3 grid((cDPROJ + 127) / 128, cM / 128);
        gemm_bf16_wmma_kernel<<<grid, 256, 0, stream>>>(u_bf, wIn_bf, nullptr, zx, cDPROJ, cDM);
    }
    // depthwise conv + silu
    {
        long long total = (long long)cM * cCONVD;
        conv_silu_kernel<<<(unsigned)((total + 255) / 256), 256, 0, stream>>>(zx, conv_w, conv_b, xconv);
    }
    // dt softplus
    dt_softplus_kernel<<<(cM * cNH + 255) / 256, 256, 0, stream>>>(zx, dt_bias, dtb);
    // ssm scan
    ssm_scan_kernel<<<cB * cNH, 256, 0, stream>>>(xconv, dtb, A_log, Dvec, ybuf);
    // gated rmsnorm -> bf16
    gate_rms_kernel<<<cM, 256, 0, stream>>>(ybuf, zx, norm_w, g_bf);
    // out_proj GEMM + residual: out[M,1024] = x + g * out_w^T
    {
        dim3 grid(cDM / 128, cM / 128);
        gemm_bf16_wmma_kernel<<<grid, 256, 0, stream>>>(g_bf, wOut_bf, x, out, cDM, cDI);
    }
    (void)in_sizes; (void)n_in; (void)out_size;
}
